// Attention_65429531787574
// MI455X (gfx1250) — matrix operand order repaired and checked by reading, not compiled, not run on hardware
//
#include <hip/hip_runtime.h>

#define B_ 64
#define S_ 2048
#define H_ 512
#define LDK 520        // padded LDS row stride (bf16 elems) for Wh chunk
#define CHUNK_N 32     // N columns staged per LDS chunk
#define NCHUNKS (H_ / CHUNK_N)

typedef __attribute__((ext_vector_type(16))) __bf16 bf16x16;
typedef __attribute__((ext_vector_type(8)))  float  floatx8;
typedef __attribute__((ext_vector_type(4)))  float  f32x4;
typedef __attribute__((ext_vector_type(4)))  int    i32x4;

__device__ __forceinline__ floatx8 wmma_bf16(bf16x16 a, bf16x16 b, floatx8 c) {
  // D = A(16x32 bf16) x B(32x16 bf16) + C(16x16 f32)
  return __builtin_amdgcn_wmma_f32_16x16x32_bf16(false, a, false, b,
                                                 (short)0, c, false, false);
}

// Build an A fragment from two 8-element f32 runs (converted to bf16).
__device__ __forceinline__ bf16x16 cvt_frag(const float* __restrict__ p0,
                                            const float* __restrict__ p1) {
  union { __bf16 h[16]; bf16x16 v; } u;
#pragma unroll
  for (int i = 0; i < 8; ++i) {
    u.h[i]     = (__bf16)p0[i];
    u.h[8 + i] = (__bf16)p1[i];
  }
  return u.v;
}

// Load a B fragment: 16 contiguous bf16 (32B) from LDS as two b128 loads.
__device__ __forceinline__ bf16x16 ld_frag_lds(const __bf16* p) {
  union { f32x4 f[2]; bf16x16 v; } u;
  u.f[0] = *(const f32x4*)(p);
  u.f[1] = *(const f32x4*)(p + 8);
  return u.v;
}

// Wait for all outstanding async global<->LDS transfers from this wave.
__device__ __forceinline__ void wait_async() {
#if __has_builtin(__builtin_amdgcn_s_wait_asynccnt)
  __builtin_amdgcn_s_wait_asynccnt(0);
#else
  asm volatile("s_wait_asynccnt 0x0" ::: "memory");
#endif
}

// Stage WhT[n0:n0+CHUNK_N][0:512] (bf16) into LDS. Uses the CDNA5 async
// global->LDS data mover when available (no VGPR round trip; ASYNCcnt).
__device__ __forceinline__ void fill_chunk(const __bf16* __restrict__ WhT,
                                           __bf16* dst, int n0, int tid) {
  for (int idx = tid; idx < CHUNK_N * (H_ / 8); idx += 256) {
    const int nl = idx >> 6;          // local column (row of WhT)
    const int kc = (idx & 63) << 3;   // 8-element (16B) k offset
    const __bf16* g = WhT + (((size_t)(n0 + nl)) << 9) + kc;
    __bf16* l = dst + nl * LDK + kc;
#if __has_builtin(__builtin_amdgcn_global_load_async_to_lds_b128)
    __builtin_amdgcn_global_load_async_to_lds_b128(
        (__attribute__((address_space(1))) i32x4*)(void*)g,
        (__attribute__((address_space(3))) i32x4*)(void*)l, 0, 0);
#else
    *(f32x4*)l = *(const f32x4*)g;
#endif
  }
}

// ---------------------------------------------------------------------------
// Kernel 1: Wh [k][n] f32  ->  WhT [n][k] bf16 (workspace)
// ---------------------------------------------------------------------------
__global__ __launch_bounds__(256) void convert_wh_kernel(
    const float* __restrict__ Wh, __bf16* __restrict__ WhT) {
  const int idx = blockIdx.x * 256 + threadIdx.x;  // 262144 total
  const int k = idx >> 9;
  const int n = idx & 511;
  WhT[((size_t)n << 9) + k] = (__bf16)Wh[idx];
}

// ---------------------------------------------------------------------------
// Kernel 2: base[b][h] = (dec[b] @ Ws)[h] + bh[h] + bs[h] + bc[h]
// ---------------------------------------------------------------------------
__global__ __launch_bounds__(256) void base_kernel(
    const float* __restrict__ dec, const float* __restrict__ Ws,
    const float* __restrict__ bh,  const float* __restrict__ bs,
    const float* __restrict__ bc,  float* __restrict__ base) {
  __shared__ float sdec[H_];
  const int b = blockIdx.x, t = threadIdx.x;
  for (int i = t; i < H_; i += 256) sdec[i] = dec[(size_t)b * H_ + i];
  __syncthreads();
  for (int h = t; h < H_; h += 256) {
    float acc = bh[h] + bs[h] + bc[h];
    for (int k = 0; k < H_; ++k) acc += sdec[k] * Ws[(size_t)k * H_ + h];
    base[(size_t)b * H_ + h] = acc;
  }
}

// ---------------------------------------------------------------------------
// Kernel 3: e[b][s] = v_w . tanh(enc@Wh + base[b] + cov[b,s]*Wc) + v_b
// Block = 8 waves, each wave owns 16 S-rows. Wh columns staged through a
// double-buffered LDS chunk filled by async global->LDS copies that overlap
// the WMMA pipeline of the current chunk.
// ---------------------------------------------------------------------------
__global__ __launch_bounds__(256) void attn_score_kernel(
    const float* __restrict__ enc,  const float* __restrict__ cov,
    const __bf16* __restrict__ WhT, const float* __restrict__ base,
    const float* __restrict__ Wc,   const float* __restrict__ vw,
    const float* __restrict__ vb,   float* __restrict__ e) {
  extern __shared__ __align__(16) char smem[];
  __bf16* sB0 = (__bf16*)smem;             // [CHUNK_N][LDK]
  __bf16* sB1 = sB0 + CHUNK_N * LDK;       // double buffer

  const int tid   = threadIdx.x;
  const int wave  = tid >> 5;
  const int lane  = tid & 31;
  const int half  = lane >> 4;   // which 16-lane half
  const int ln    = lane & 15;
  const int b     = blockIdx.x >> 4;
  const int s0    = (blockIdx.x & 15) << 7;  // 128-row S tile
  const int sBase = s0 + wave * 16;

  // Kick off the first chunk's async fill before doing register work.
  fill_chunk(WhT, sB0, 0, tid);

  // A fragments for this wave's 16x512 tile, converted f32->bf16 once.
  // A layout (matches B): lane half 0 holds K 0..15, half 1 holds K 16..31.
  bf16x16 afrag[16];
  {
    const float* arow = enc + ((size_t)b * S_ + sBase + ln) * H_;
#pragma unroll
    for (int kc = 0; kc < 16; ++kc) {
      const int k0 = kc * 32 + half * 16;
      afrag[kc] = cvt_frag(arow + k0, arow + k0 + 8);
    }
  }

  // Coverage values for this lane's 8 D-rows (rows r + 8*half).
  float cr[8];
#pragma unroll
  for (int r = 0; r < 8; ++r)
    cr[r] = cov[(size_t)b * S_ + sBase + half * 8 + r];

  float eacc[8] = {0.f, 0.f, 0.f, 0.f, 0.f, 0.f, 0.f, 0.f};

  wait_async();
  __syncthreads();  // chunk 0 visible to all waves

  for (int c = 0; c < NCHUNKS; ++c) {
    const int n0 = c * CHUNK_N;
    __bf16* cur = (c & 1) ? sB1 : sB0;
    // Overlap: async-fill the next chunk into the other buffer while the
    // WMMA pipeline consumes the current one.
    if (c + 1 < NCHUNKS)
      fill_chunk(WhT, (c & 1) ? sB0 : sB1, n0 + CHUNK_N, tid);

#pragma unroll
    for (int nt = 0; nt < 2; ++nt) {
      floatx8 acc = {0.f, 0.f, 0.f, 0.f, 0.f, 0.f, 0.f, 0.f};
      // B layout: lane ln = column, half selects K 0..15 vs 16..31 per step.
      const __bf16* pB = cur + (nt * 16 + ln) * LDK + 16 * half;
#pragma unroll
      for (int kc = 0; kc < 16; ++kc) {
        bf16x16 bfrag = ld_frag_lds(pB + kc * 32);
        acc = wmma_bf16(afrag[kc], bfrag, acc);
      }
      // Fused epilogue on the D fragment: +base +cov*Wc, tanh, *v_w.
      const int n = n0 + nt * 16 + ln;     // this lane's column
      const float bb  = base[(b << 9) + n];
      const float wc  = Wc[n];
      const float vwn = vw[n];
#pragma unroll
      for (int r = 0; r < 8; ++r) {
        const float val = acc[r] + bb + cr[r] * wc;
        eacc[r] += tanhf(val) * vwn;
      }
    }

    if (c + 1 < NCHUNKS) {
      wait_async();     // next chunk's copies (this wave's share) landed
      __syncthreads();  // ...and every other wave's share too
    }
  }

  // Sum over the 16 columns held across lanes (stays within each half-wave).
#pragma unroll
  for (int off = 1; off < 16; off <<= 1) {
#pragma unroll
    for (int r = 0; r < 8; ++r)
      eacc[r] += __shfl_xor(eacc[r], off, 32);
  }
  if (ln == 0) {
    const float vbv = vb[0];
#pragma unroll
    for (int r = 0; r < 8; ++r)
      e[(size_t)b * S_ + sBase + half * 8 + r] = eacc[r] + vbv;
  }
}

// ---------------------------------------------------------------------------
// Kernel 4: softmax over S per batch; a_t and coverage + a_t.
// ---------------------------------------------------------------------------
__global__ __launch_bounds__(256) void softmax_cov_kernel(
    const float* __restrict__ e, const float* __restrict__ cov,
    float* __restrict__ out) {
  __shared__ float red[256];
  const int b = blockIdx.x, t = threadIdx.x;
  float v[8];
  float m = -1e30f;
#pragma unroll
  for (int i = 0; i < 8; ++i) {
    v[i] = e[(size_t)b * S_ + i * 256 + t];
    m = fmaxf(m, v[i]);
  }
  red[t] = m;
  __syncthreads();
  for (int s = 128; s > 0; s >>= 1) {
    if (t < s) red[t] = fmaxf(red[t], red[t + s]);
    __syncthreads();
  }
  m = red[0];
  __syncthreads();
  float sum = 0.f;
#pragma unroll
  for (int i = 0; i < 8; ++i) { v[i] = __expf(v[i] - m); sum += v[i]; }
  red[t] = sum;
  __syncthreads();
  for (int s = 128; s > 0; s >>= 1) {
    if (t < s) red[t] += red[t + s];
    __syncthreads();
  }
  const float inv = 1.f / red[0];
#pragma unroll
  for (int i = 0; i < 8; ++i) {
    const size_t idx = (size_t)b * S_ + i * 256 + t;
    const float a = v[i] * inv;
    out[idx] = a;                              // a_t
    out[(size_t)B_ * S_ + idx] = cov[idx] + a; // sum_coverage
  }
}

// ---------------------------------------------------------------------------
extern "C" void kernel_launch(void* const* d_in, const int* in_sizes, int n_in,
                              void* d_out, int out_size, void* d_ws, size_t ws_size,
                              hipStream_t stream) {
  const float* enc = (const float*)d_in[0];
  const float* dec = (const float*)d_in[1];
  const float* cov = (const float*)d_in[2];
  const float* Wh  = (const float*)d_in[3];
  const float* bh  = (const float*)d_in[4];
  const float* Ws  = (const float*)d_in[5];
  const float* bs  = (const float*)d_in[6];
  const float* Wc  = (const float*)d_in[7];  // [1,H] -> flat [H]
  const float* bc  = (const float*)d_in[8];
  const float* vw  = (const float*)d_in[9];
  const float* vb  = (const float*)d_in[10];
  float* out = (float*)d_out;

  // Workspace layout: WhT bf16 (512KB) | base f32 (128KB) | e f32 (512KB)
  char* ws = (char*)d_ws;
  __bf16* WhT  = (__bf16*)ws;
  float*  base = (float*)(ws + (size_t)H_ * H_ * 2);
  float*  e    = (float*)(ws + (size_t)H_ * H_ * 2 + (size_t)B_ * H_ * 4);

  convert_wh_kernel<<<(H_ * H_) / 256, 256, 0, stream>>>(Wh, WhT);
  base_kernel<<<B_, 256, 0, stream>>>(dec, Ws, bh, bs, bc, base);
  attn_score_kernel<<<B_ * (S_ / 128), 256,
                      2 * CHUNK_N * LDK * sizeof(__bf16), stream>>>(
      enc, cov, WhT, base, Wc, vw, vb, e);
  softmax_cov_kernel<<<B_, 256, 0, stream>>>(e, cov, out);
}
